// GCN_41016937677072
// MI455X (gfx1250) — matrix operand order repaired and checked by reading, not compiled, not run on hardware
//
#include <hip/hip_runtime.h>
#include <hip/hip_bf16.h>
#include <math.h>

// ---------------------------------------------------------------------------
// GraphConv x3 + log_softmax for MI455X (gfx1250, wave32, WMMA bf16->f32)
//   - segment_sum via native f32 atomics (L2 atomic units)
//   - GEMMs: LDS-staged bf16 A panels shared by 8 waves/block,
//            bf16 [N,K]-transposed weights -> b128 fragment loads,
//            v_wmma_f32_16x16x32_bf16 accumulation in f32
// ---------------------------------------------------------------------------

#define N_NODES 100000
#define N_EDGES 800000
#define IN_DIM  128
#define HID     128
#define EMB     256
#define N_CLS   16

typedef __attribute__((ext_vector_type(16))) __bf16 v16bf;
typedef __attribute__((ext_vector_type(8)))  float  v8f;

union U4BF8 { uint4 u; __bf16 h[8]; };
union U2BF4 { uint2 u; __bf16 h[4]; };

// ---------------------------------------------------------------------------
// Zero-fill (float4 vectorized)
// ---------------------------------------------------------------------------
__global__ void zero_kernel(float4* __restrict__ p, long long n4) {
    long long i = (long long)blockIdx.x * blockDim.x + threadIdx.x;
    if (i < n4) p[i] = make_float4(0.f, 0.f, 0.f, 0.f);
}

// ---------------------------------------------------------------------------
// Weight convert + transpose: Wt[n*K + k] = bf16( W[k*N + n] )
// ---------------------------------------------------------------------------
__global__ void wt_convert_kernel(const float* __restrict__ W,
                                  __bf16* __restrict__ Wt, int K, int N) {
    int i = blockIdx.x * blockDim.x + threadIdx.x;
    if (i >= N * K) return;
    int n = i / K;
    int k = i - n * K;
    Wt[i] = (__bf16)W[(long long)k * N + n];
}

// ---------------------------------------------------------------------------
// Scatter-add: AGG[dst[e], f] += X[src[e], f]   (D = 1<<shift features)
// ---------------------------------------------------------------------------
__global__ void scatter_add_kernel(const float* __restrict__ X,
                                   const int* __restrict__ src,
                                   const int* __restrict__ dst,
                                   float* __restrict__ AGG,
                                   int shift) {
    long long i = (long long)blockIdx.x * blockDim.x + threadIdx.x;
    long long total = (long long)N_EDGES << shift;
    if (i >= total) return;
    int e = (int)(i >> shift);
    int f = (int)(i & ((1 << shift) - 1));
    int s = src[e];
    int d = dst[e];
    float v = X[((long long)s << shift) + f];
    unsafeAtomicAdd(&AGG[((long long)d << shift) + f], v);
}

// ---------------------------------------------------------------------------
// Fragment helpers (ISA 7.12.2 layouts, wave32)
//   A (16x32 bf16): lane<16: row=lane,    K = k0 + {0..7, 16..23}
//                   lane>=16: row=lane-16, K = k0 + {8..15, 24..31}
//   B (32x16 bf16): col=lane&15; lane<16: K = k0+{0..7,16..23}; else K = k0+{8..15,24..31}
// ---------------------------------------------------------------------------
__device__ __forceinline__ v16bf load_a_frag_lds(const __bf16* sA, int Kp,
                                                 int k0, int laneLo, int laneHi) {
    const __bf16* ap = sA + laneLo * Kp + k0 + laneHi * 8;
    U4BF8 a0, a1;
    a0.u = *(const uint4*)ap;          // K = k0 + laneHi*8 + {0..7}
    a1.u = *(const uint4*)(ap + 16);   // K = k0 + 16 + laneHi*8 + {0..7}
    v16bf a;
#pragma unroll
    for (int e = 0; e < 8; ++e) { a[e] = a0.h[e]; a[e + 8] = a1.h[e]; }
    return a;
}

__device__ __forceinline__ v16bf load_b_frag(const __bf16* __restrict__ Wt, int K,
                                             int n0, int k0, int laneLo, int laneHi) {
    const __bf16* bp = Wt + (long long)(n0 + laneLo) * K + k0 + laneHi * 8;
    U4BF8 b0, b1;
    b0.u = *(const uint4*)bp;          // K = k0 + laneHi*8 + {0..7}
    b1.u = *(const uint4*)(bp + 16);   // K = k0 + 16 + laneHi*8 + {0..7}
    v16bf b;
#pragma unroll
    for (int e = 0; e < 8; ++e) { b[e] = b0.h[e]; b[e + 8] = b1.h[e]; }
    return b;
}

// ---------------------------------------------------------------------------
// Fused GraphConv GEMM with LDS staging:
//   Out = act( AGG@Wrel + Xroot@Wroot + bias )
// Block = 256 threads = 8 waves. Block handles one m-tile (16 rows) and a
// group of 8 consecutive n-tiles; the 16xK A panels (both terms) are staged
// once in LDS as bf16 and shared by all waves. N/16 must be a multiple of 8.
// ---------------------------------------------------------------------------
template <int K, int RELU>
__global__ __launch_bounds__(256)
void graphconv_gemm_lds_kernel(const float* __restrict__ AGG,
                               const float* __restrict__ Xroot,
                               const __bf16* __restrict__ WrelT,   // [N,K] bf16
                               const __bf16* __restrict__ WrootT,  // [N,K] bf16
                               const float* __restrict__ bias,
                               float* __restrict__ Out,
                               int N, int nGroups) {
    constexpr int Kp = K + 8;                 // +16B row pad: bank rotation
    __shared__ __bf16 sA[16 * Kp];            // rel term panel (from AGG)
    __shared__ __bf16 sX[16 * Kp];            // root term panel (from Xroot)

    const int mTile  = blockIdx.x / nGroups;
    const int ngroup = blockIdx.x - mTile * nGroups;
    const int m0 = mTile << 4;

    // ---- cooperative stage: 16 rows x K f32 -> bf16 LDS (both panels) ----
    const float4* Ag = (const float4*)(AGG   + (long long)m0 * K);
    const float4* Xg = (const float4*)(Xroot + (long long)m0 * K);
    const int n4 = 16 * K / 4;
    for (int i = threadIdx.x; i < n4; i += 256) {
        int row = i / (K / 4);
        int c4  = i - row * (K / 4);
        float4 va = Ag[i];
        float4 vx = Xg[i];
        U2BF4 pa, px;
        pa.h[0] = (__bf16)va.x; pa.h[1] = (__bf16)va.y;
        pa.h[2] = (__bf16)va.z; pa.h[3] = (__bf16)va.w;
        px.h[0] = (__bf16)vx.x; px.h[1] = (__bf16)vx.y;
        px.h[2] = (__bf16)vx.z; px.h[3] = (__bf16)vx.w;
        *(uint2*)&sA[row * Kp + c4 * 4] = pa.u;
        *(uint2*)&sX[row * Kp + c4 * 4] = px.u;
    }
    __syncthreads();

    // ---- per-wave 16x16 tile ----
    const int lane   = threadIdx.x & 31;
    const int wave   = threadIdx.x >> 5;
    const int laneLo = lane & 15;
    const int laneHi = lane >> 4;
    const int nTile  = ngroup * 8 + wave;     // always < N/16 (exact division)
    const int n0     = nTile << 4;

    v8f acc = {};
#pragma unroll
    for (int k0 = 0; k0 < K; k0 += 32) {
        v16bf aA = load_a_frag_lds(sA, Kp, k0, laneLo, laneHi);
        v16bf bR = load_b_frag(WrelT, K, n0, k0, laneLo, laneHi);
        acc = __builtin_amdgcn_wmma_f32_16x16x32_bf16(false, aA, false, bR,
                                                      (short)0, acc, false, false);
        v16bf aX = load_a_frag_lds(sX, Kp, k0, laneLo, laneHi);
        v16bf bQ = load_b_frag(WrootT, K, n0, k0, laneLo, laneHi);
        acc = __builtin_amdgcn_wmma_f32_16x16x32_bf16(false, aX, false, bQ,
                                                      (short)0, acc, false, false);
    }

    // C/D layout: VGPR r -> row m0 + r + laneHi*8, col n0 + laneLo
    const int col   = n0 + laneLo;
    const int rbase = m0 + (laneHi << 3);
    const float bv  = bias[col];
#pragma unroll
    for (int r = 0; r < 8; ++r) {
        float v = acc[r] + bv;
        if (RELU) v = fmaxf(v, 0.f);
        Out[(long long)(rbase + r) * N + col] = v;
    }
}

// ---------------------------------------------------------------------------
// Tail GEMM (layer 3: N=16, K=256): one wave per 16x16 tile, no LDS.
// A fragments read f32 + convert; B from bf16 transposed weights.
// ---------------------------------------------------------------------------
__global__ __launch_bounds__(256)
void gemm_tail_kernel(const float* __restrict__ AGG,
                      const float* __restrict__ Xroot,
                      const __bf16* __restrict__ WrelT,
                      const __bf16* __restrict__ WrootT,
                      const float* __restrict__ bias,
                      float* __restrict__ Out,
                      int M, int N, int K) {
    const int lane = threadIdx.x & 31;
    const int wave = threadIdx.x >> 5;
    const long long totalTiles = (long long)(M >> 4);   // nTilesN == 1
    const long long tile = (long long)blockIdx.x * (blockDim.x >> 5) + wave;
    if (tile >= totalTiles) return;   // wave-uniform: EXEC all-1s inside

    const int m0 = (int)tile << 4;
    const int laneLo = lane & 15;
    const int laneHi = lane >> 4;

    v8f acc = {};
    for (int k0 = 0; k0 < K; k0 += 32) {
        // A fragments from f32 global, cvt to bf16
        const float* arA = AGG   + (long long)(m0 + laneLo) * K + k0 + laneHi * 8;
        const float* arX = Xroot + (long long)(m0 + laneLo) * K + k0 + laneHi * 8;
        v16bf aA, aX;
#pragma unroll
        for (int e = 0; e < 8; ++e) {
            aA[e]     = (__bf16)arA[e];
            aA[e + 8] = (__bf16)arA[16 + e];
            aX[e]     = (__bf16)arX[e];
            aX[e + 8] = (__bf16)arX[16 + e];
        }
        v16bf bR = load_b_frag(WrelT,  K, 0, k0, laneLo, laneHi);
        v16bf bQ = load_b_frag(WrootT, K, 0, k0, laneLo, laneHi);
        acc = __builtin_amdgcn_wmma_f32_16x16x32_bf16(false, aA, false, bR,
                                                      (short)0, acc, false, false);
        acc = __builtin_amdgcn_wmma_f32_16x16x32_bf16(false, aX, false, bQ,
                                                      (short)0, acc, false, false);
    }

    const int col   = laneLo;
    const int rbase = m0 + (laneHi << 3);
    const float bv  = bias[col];
#pragma unroll
    for (int r = 0; r < 8; ++r) {
        Out[(long long)(rbase + r) * N + col] = acc[r] + bv;
    }
}

// ---------------------------------------------------------------------------
// In-place log_softmax over rows of 16
// ---------------------------------------------------------------------------
__global__ void log_softmax16_kernel(float* __restrict__ Out, int M) {
    int row = blockIdx.x * blockDim.x + threadIdx.x;
    if (row >= M) return;
    float v[N_CLS];
    float mx = -INFINITY;
#pragma unroll
    for (int i = 0; i < N_CLS; ++i) {
        v[i] = Out[(long long)row * N_CLS + i];
        mx = fmaxf(mx, v[i]);
    }
    float s = 0.f;
#pragma unroll
    for (int i = 0; i < N_CLS; ++i) s += __expf(v[i] - mx);
    float lse = mx + __logf(s);
#pragma unroll
    for (int i = 0; i < N_CLS; ++i) Out[(long long)row * N_CLS + i] = v[i] - lse;
}

// ---------------------------------------------------------------------------
// Host-side orchestration
// ---------------------------------------------------------------------------
static inline void launch_zero(float* p, long long nFloats, hipStream_t s) {
    long long n4 = nFloats >> 2;
    int blk = 256;
    long long grid = (n4 + blk - 1) / blk;
    zero_kernel<<<(unsigned)grid, blk, 0, s>>>((float4*)p, n4);
}

extern "C" void kernel_launch(void* const* d_in, const int* in_sizes, int n_in,
                              void* d_out, int out_size, void* d_ws, size_t ws_size,
                              hipStream_t stream) {
    const float* x      = (const float*)d_in[0];
    const int*   eidx   = (const int*)d_in[1];
    const float* Wrel1  = (const float*)d_in[2];
    const float* Wroot1 = (const float*)d_in[3];
    const float* b1     = (const float*)d_in[4];
    const float* Wrel2  = (const float*)d_in[5];
    const float* Wroot2 = (const float*)d_in[6];
    const float* b2     = (const float*)d_in[7];
    const float* Wrel3  = (const float*)d_in[8];
    const float* Wroot3 = (const float*)d_in[9];
    const float* b3     = (const float*)d_in[10];

    const int* src = eidx;               // edge_index[0]
    const int* dst = eidx + N_EDGES;     // edge_index[1]

    float* ws  = (float*)d_ws;
    float* AGG = ws;                                          // 100000*256 f32
    float* H1  = AGG + (long long)N_NODES * EMB;              // 100000*128 f32
    float* H2  = H1  + (long long)N_NODES * HID;              // 100000*256 f32
    __bf16* WT = (__bf16*)(H2 + (long long)N_NODES * EMB);    // bf16 weights
    __bf16* Wrel1T  = WT;                                     // [128,128]
    __bf16* Wroot1T = Wrel1T  + IN_DIM * HID;
    __bf16* Wrel2T  = Wroot1T + IN_DIM * HID;                 // [256,128]
    __bf16* Wroot2T = Wrel2T  + HID * EMB;
    __bf16* Wrel3T  = Wroot2T + HID * EMB;                    // [16,256]
    __bf16* Wroot3T = Wrel3T  + EMB * N_CLS;
    float* outF = (float*)d_out;                              // 100000*16 f32

    const int blk = 256;

    // ---------------- weight convert + transpose (tiny) --------------------
    wt_convert_kernel<<<(IN_DIM * HID + blk - 1) / blk, blk, 0, stream>>>(Wrel1,  Wrel1T,  IN_DIM, HID);
    wt_convert_kernel<<<(IN_DIM * HID + blk - 1) / blk, blk, 0, stream>>>(Wroot1, Wroot1T, IN_DIM, HID);
    wt_convert_kernel<<<(HID * EMB   + blk - 1) / blk, blk, 0, stream>>>(Wrel2,  Wrel2T,  HID, EMB);
    wt_convert_kernel<<<(HID * EMB   + blk - 1) / blk, blk, 0, stream>>>(Wroot2, Wroot2T, HID, EMB);
    wt_convert_kernel<<<(EMB * N_CLS + blk - 1) / blk, blk, 0, stream>>>(Wrel3,  Wrel3T,  EMB, N_CLS);
    wt_convert_kernel<<<(EMB * N_CLS + blk - 1) / blk, blk, 0, stream>>>(Wroot3, Wroot3T, EMB, N_CLS);

    // ---------------- Layer 1: GraphConv(128 -> 128) + ReLU ----------------
    launch_zero(AGG, (long long)N_NODES * IN_DIM, stream);
    {
        long long total = (long long)N_EDGES << 7;
        scatter_add_kernel<<<(unsigned)((total + blk - 1) / blk), blk, 0, stream>>>(
            x, src, dst, AGG, 7);
    }
    {
        int nGroups = (HID / 16) / 8;                     // 1
        unsigned grid = (unsigned)(N_NODES / 16) * nGroups;
        graphconv_gemm_lds_kernel<IN_DIM, 1><<<grid, blk, 0, stream>>>(
            AGG, x, Wrel1T, Wroot1T, b1, H1, HID, nGroups);
    }

    // ---------------- Layer 2: GraphConv(128 -> 256) + ReLU ----------------
    launch_zero(AGG, (long long)N_NODES * HID, stream);
    {
        long long total = (long long)N_EDGES << 7;
        scatter_add_kernel<<<(unsigned)((total + blk - 1) / blk), blk, 0, stream>>>(
            H1, src, dst, AGG, 7);
    }
    {
        int nGroups = (EMB / 16) / 8;                     // 2
        unsigned grid = (unsigned)(N_NODES / 16) * nGroups;
        graphconv_gemm_lds_kernel<HID, 1><<<grid, blk, 0, stream>>>(
            AGG, H1, Wrel2T, Wroot2T, b2, H2, EMB, nGroups);
    }

    // ---------------- Layer 3: GraphConv(256 -> 16) ------------------------
    launch_zero(AGG, (long long)N_NODES * EMB, stream);
    {
        long long total = (long long)N_EDGES << 8;
        scatter_add_kernel<<<(unsigned)((total + blk - 1) / blk), blk, 0, stream>>>(
            H2, src, dst, AGG, 8);
    }
    {
        long long tiles = N_NODES / 16;                   // 6250
        unsigned grid = (unsigned)((tiles + 7) / 8);
        gemm_tail_kernel<<<grid, blk, 0, stream>>>(
            AGG, H2, Wrel3T, Wroot3T, b3, outF, N_NODES, N_CLS, EMB);
    }

    // ---------------- log_softmax over 16 classes --------------------------
    log_softmax16_kernel<<<(N_NODES + blk - 1) / blk, blk, 0, stream>>>(outF, N_NODES);
}
